// EnhancedMambaFormerBlock_23957327577741
// MI455X (gfx1250) — matrix operand order repaired and checked by reading, not compiled, not run on hardware
//
#include <hip/hip_runtime.h>
#include <math.h>

// ---------------------------------------------------------------------------
// EnhancedMambaFormerBlock for MI455X (gfx1250, wave32, WMMA f16->f32).
// All GEMMs use one batched WMMA kernel (v_wmma_f32_16x16x32_f16).
// B operands are pre-packed to f16 [N,K]; fragments are contiguous 32B
// vector loads. A stays f32 (two 32B vector loads + packed cvt).
// K loop is branchless and manually unrolled x2 with disjoint fragment
// register sets so loads overlap WMMA and the f16 WMMA->VALU WAR hazard
// (4 co-exec NOPs) is avoided. All K are multiples of 64 here.
// Wave tile 16x64 = 4 accumulators sharing one A fragment; 8 waves -> 64x128.
// Softmax writes attention weights straight into d_out (part of the output
// tuple) and the P*V GEMM consumes them from there.
// Workspace: ~248 MB (f32 activations + f16 packed operands).
// ---------------------------------------------------------------------------

typedef __attribute__((ext_vector_type(16))) _Float16 v16h;
typedef __attribute__((ext_vector_type(8)))  float    v8f;

#define LN_EPS 1e-5f

__device__ __forceinline__ float act_apply(float v, int act) {
  switch (act) {
    case 1: return v / (1.f + __expf(-v));                       // silu
    case 2: {                                                    // gelu (tanh approx, jax default)
      float u = 0.7978845608028654f * (v + 0.044715f * v * v * v);
      return 0.5f * v * (1.f + tanhf(u));
    }
    case 3: return (v > 20.f) ? v : log1pf(__expf(v));           // softplus
    case 4: return 1.f / (1.f + __expf(-v));                     // sigmoid
    default: return v;
  }
}

__device__ __forceinline__ v16h cvt_frag(const v8f lo, const v8f hi) {
  v16h af;
  #pragma unroll
  for (int e = 0; e < 8; ++e) { af[e] = (_Float16)lo[e]; af[8 + e] = (_Float16)hi[e]; }
  return af;
}

__device__ __forceinline__ void store_tile(
    float* __restrict__ Cb, long ldc, const v8f acc,
    int row0, int hf, int cn, int N, float scale, int act,
    const float* __restrict__ bias, const float* __restrict__ resid)
{
  if (cn >= N) return;
  const float bv = bias ? bias[cn] : 0.f;
  #pragma unroll
  for (int r = 0; r < 8; ++r) {
    const int rr = row0 + r + hf * 8;   // M is always a multiple of 16 here
    float v = acc[r] * scale + bv;
    v = act_apply(v, act);
    if (resid) v += resid[(long)rr * ldc + cn];
    Cb[(long)rr * ldc + cn] = v;
  }
}

// ---------------------------------------------------------------------------
// Batched GEMM:  C[z] = act( scale * A[z] @ Bt[z]^T + bias ) + resid
//   A : [M,K] row-major f32 ; Bt: [N,K] row-major f16 (pre-packed)
//   z -> (b,h) = (z/Hdim, z%Hdim) with independent strides.
// K multiple of 64; leading dims multiples of 32 elements (32B alignment).
// ---------------------------------------------------------------------------
__global__ __launch_bounds__(256) void gemm_wmma_kernel(
    const float* __restrict__ A, long lda, long sAb, long sAh,
    const _Float16* __restrict__ Bt, long ldb, long sBb, long sBh,
    float* __restrict__ C, long ldc, long sCb, long sCh,
    const float* __restrict__ bias, const float* __restrict__ resid,
    int M, int N, int K, int Hdim, float scale, int act)
{
  const int z  = blockIdx.z;
  const int bb = z / Hdim, hh = z - bb * Hdim;
  const float*    Ab  = A  + (long)bb * sAb + (long)hh * sAh;
  const _Float16* Btb = Bt + (long)bb * sBb + (long)hh * sBh;
  float*          Cb  = C  + (long)bb * sCb + (long)hh * sCh;

  const int wave = threadIdx.x >> 5;
  const int lane = threadIdx.x & 31;
  const int wm = wave >> 1, wn = wave & 1;
  const int row0 = blockIdx.y * 64 + wm * 16;
  const int colB = blockIdx.x * 128 + wn * 64;
  if (row0 >= M || colB >= N) return;   // no barriers in this kernel -> safe

  const int ml = lane & 15;   // row (A) / col (B) within 16-tile
  const int hf = lane >> 4;   // lane-half selects K sub-block

  // Column indices, clamped so loads never fault; stores are guarded once.
  const int cn0 = colB + 0 * 16 + ml;
  const int cn1 = colB + 1 * 16 + ml;
  const int cn2 = colB + 2 * 16 + ml;
  const int cn3 = colB + 3 * 16 + ml;
  const int cc0 = cn0 < N ? cn0 : N - 1;
  const int cc1 = cn1 < N ? cn1 : N - 1;
  const int cc2 = cn2 < N ? cn2 : N - 1;
  const int cc3 = cn3 < N ? cn3 : N - 1;

  // Hoisted base pointers (lane-half K offset folded in).
  const float*    ap  = Ab + (long)(row0 + ml) * lda + 16 * hf;
  const _Float16* bp0 = Btb + (long)cc0 * ldb + 16 * hf;
  const _Float16* bp1 = Btb + (long)cc1 * ldb + 16 * hf;
  const _Float16* bp2 = Btb + (long)cc2 * ldb + 16 * hf;
  const _Float16* bp3 = Btb + (long)cc3 * ldb + 16 * hf;

  v8f acc0 = {}, acc1 = {}, acc2 = {}, acc3 = {};

  // Two K-steps per iteration with disjoint fragment registers: all 12
  // vector loads issue before the 8 WMMAs, hiding latency and avoiding the
  // WMMA->VALU WAR hazard on fragment registers.
  for (int k0 = 0; k0 < K; k0 += 64) {
    const v8f lo0 = *(const v8f*)(ap + k0);
    const v8f hi0 = *(const v8f*)(ap + k0 + 8);
    const v8f lo1 = *(const v8f*)(ap + k0 + 32);
    const v8f hi1 = *(const v8f*)(ap + k0 + 40);
    const v16h b00 = *(const v16h*)(bp0 + k0);
    const v16h b01 = *(const v16h*)(bp1 + k0);
    const v16h b02 = *(const v16h*)(bp2 + k0);
    const v16h b03 = *(const v16h*)(bp3 + k0);
    const v16h b10 = *(const v16h*)(bp0 + k0 + 32);
    const v16h b11 = *(const v16h*)(bp1 + k0 + 32);
    const v16h b12 = *(const v16h*)(bp2 + k0 + 32);
    const v16h b13 = *(const v16h*)(bp3 + k0 + 32);
    const v16h af0 = cvt_frag(lo0, hi0);
    const v16h af1 = cvt_frag(lo1, hi1);
    acc0 = __builtin_amdgcn_wmma_f32_16x16x32_f16(false, af0, false, b00, (short)0, acc0, false, false);
    acc1 = __builtin_amdgcn_wmma_f32_16x16x32_f16(false, af0, false, b01, (short)0, acc1, false, false);
    acc2 = __builtin_amdgcn_wmma_f32_16x16x32_f16(false, af0, false, b02, (short)0, acc2, false, false);
    acc3 = __builtin_amdgcn_wmma_f32_16x16x32_f16(false, af0, false, b03, (short)0, acc3, false, false);
    acc0 = __builtin_amdgcn_wmma_f32_16x16x32_f16(false, af1, false, b10, (short)0, acc0, false, false);
    acc1 = __builtin_amdgcn_wmma_f32_16x16x32_f16(false, af1, false, b11, (short)0, acc1, false, false);
    acc2 = __builtin_amdgcn_wmma_f32_16x16x32_f16(false, af1, false, b12, (short)0, acc2, false, false);
    acc3 = __builtin_amdgcn_wmma_f32_16x16x32_f16(false, af1, false, b13, (short)0, acc3, false, false);
  }

  // Epilogue. C layout: VGPR r -> row row0 + r + 8*hf, col = colB + 16t + ml.
  store_tile(Cb, ldc, acc0, row0, hf, cn0, N, scale, act, bias, resid);
  store_tile(Cb, ldc, acc1, row0, hf, cn1, N, scale, act, bias, resid);
  store_tile(Cb, ldc, acc2, row0, hf, cn2, N, scale, act, bias, resid);
  store_tile(Cb, ldc, acc3, row0, hf, cn3, N, scale, act, bias, resid);
}

// ---------------------------------------------------------------------------
// Batched tiled transpose + f32->f16 convert:  out[N,K] = (f16) in[K,N]^T.
// ---------------------------------------------------------------------------
__global__ __launch_bounds__(256) void transpose_f16_kernel(
    const float* __restrict__ in, long ldin, long sIb, long sIh,
    _Float16* __restrict__ out, long ldout, long sOb, long sOh,
    int R, int Cc, int Hdim)
{
  __shared__ float tile[32][33];
  const int z = blockIdx.z;
  const int bb = z / Hdim, hh = z - bb * Hdim;
  const float* ip = in + (long)bb * sIb + (long)hh * sIh;
  _Float16*    op = out + (long)bb * sOb + (long)hh * sOh;
  const int c0 = blockIdx.x * 32, r0 = blockIdx.y * 32;
  const int tx = threadIdx.x & 31, ty = threadIdx.x >> 5;
  #pragma unroll
  for (int i = 0; i < 32; i += 8) {
    int r = r0 + ty + i, c = c0 + tx;
    tile[ty + i][tx] = (r < R && c < Cc) ? ip[(long)r * ldin + c] : 0.f;
  }
  __syncthreads();
  #pragma unroll
  for (int i = 0; i < 32; i += 8) {
    int rO = c0 + ty + i;   // output row = input col
    int cO = r0 + tx;       // output col = input row
    if (rO < Cc && cO < R) op[(long)rO * ldout + cO] = (_Float16)tile[tx][ty + i];
  }
}

// Straight f32 -> f16 convert (for operands already N-major, e.g. K matrix).
__global__ void convert_f16_kernel(const float* __restrict__ in,
                                   _Float16* __restrict__ out, long n) {
  long i = (long)blockIdx.x * blockDim.x + threadIdx.x;
  if (i < n) out[i] = (_Float16)in[i];
}

// ---------------------------------------------------------------------------
// LayerNorm over last dim D (one block per row).
// ---------------------------------------------------------------------------
__global__ __launch_bounds__(256) void layernorm_kernel(
    const float* __restrict__ in, const float* __restrict__ g,
    const float* __restrict__ b, float* __restrict__ out, int D)
{
  __shared__ float sh1[256], sh2[256];
  const long row = blockIdx.x;
  const float* x = in + row * (long)D;
  const int t = threadIdx.x;
  float s = 0.f, s2 = 0.f;
  for (int i = t; i < D; i += 256) { float v = x[i]; s += v; s2 += v * v; }
  sh1[t] = s; sh2[t] = s2; __syncthreads();
  for (int off = 128; off > 0; off >>= 1) {
    if (t < off) { sh1[t] += sh1[t + off]; sh2[t] += sh2[t + off]; }
    __syncthreads();
  }
  const float mu = sh1[0] / D;
  const float var = sh2[0] / D - mu * mu;
  const float rstd = rsqrtf(var + LN_EPS);
  for (int i = t; i < D; i += 256)
    out[row * (long)D + i] = (x[i] - mu) * rstd * g[i] + b[i];
}

// ---------------------------------------------------------------------------
// Row softmax, in place (scores pre-scaled by 1/sqrt(dk) in the GEMM).
// ---------------------------------------------------------------------------
__global__ __launch_bounds__(256) void softmax_kernel(float* __restrict__ p, int cols)
{
  __shared__ float sh[256];
  const long row = blockIdx.x;
  float* pr = p + row * (long)cols;
  const int t = threadIdx.x;
  float mx = -3.4e38f;
  for (int i = t; i < cols; i += 256) mx = fmaxf(mx, pr[i]);
  sh[t] = mx; __syncthreads();
  for (int off = 128; off > 0; off >>= 1) {
    if (t < off) sh[t] = fmaxf(sh[t], sh[t + off]);
    __syncthreads();
  }
  const float m = sh[0]; __syncthreads();
  float s = 0.f;
  for (int i = t; i < cols; i += 256) { float e = __expf(pr[i] - m); pr[i] = e; s += e; }
  sh[t] = s; __syncthreads();
  for (int off = 128; off > 0; off >>= 1) {
    if (t < off) sh[t] += sh[t + off];
    __syncthreads();
  }
  const float inv = 1.f / sh[0];
  for (int i = t; i < cols; i += 256) pr[i] *= inv;
}

// ---------------------------------------------------------------------------
// Causal depthwise conv1d (width 4) + SiLU. xm = first DI cols of in_proj out.
// ---------------------------------------------------------------------------
__global__ __launch_bounds__(256) void conv_silu_kernel(
    const float* __restrict__ xm, long ldxm,
    const float* __restrict__ w, const float* __restrict__ cb,
    float* __restrict__ xc, int L, int DI, long total)
{
  long idx = (long)blockIdx.x * blockDim.x + threadIdx.x;
  if (idx >= total) return;
  int d = (int)(idx % DI);
  long tk = idx / DI;
  int l = (int)(tk % L);
  long bi = tk / L;
  float acc = cb[d];
  #pragma unroll
  for (int k = 0; k < 4; ++k) {
    int li = l + k - 3;
    if (li >= 0) acc += w[d * 4 + k] * xm[(bi * L + li) * ldxm + d];
  }
  xc[idx] = act_apply(acc, 1);
}

// ---------------------------------------------------------------------------
// Selective scan: one thread per (batch, channel), 16 SSM states in regs.
// B/C per (b,l) are uniform across a wave's lanes -> broadcast loads.
// ---------------------------------------------------------------------------
__global__ __launch_bounds__(256) void scan_kernel(
    const float* __restrict__ u, const float* __restrict__ delta,
    const float* __restrict__ BC, const float* __restrict__ A_log,
    const float* __restrict__ Dp, float* __restrict__ y, int L, int DI)
{
  const int tid = blockIdx.x * blockDim.x + threadIdx.x;
  const int d = tid % DI;
  const int bi = tid / DI;
  float A[16], h[16];
  #pragma unroll
  for (int s = 0; s < 16; ++s) { A[s] = -__expf(A_log[s]); h[s] = 0.f; }
  const float dpv = Dp[d];
  for (int l = 0; l < L; ++l) {
    const long tk = (long)bi * L + l;
    const float uv = u[tk * DI + d];
    const float dl = delta[tk * DI + d];
    const float* bc = BC + tk * 32;
    float acc = 0.f;
    #pragma unroll
    for (int s = 0; s < 16; ++s) {
      float dA = __expf(dl * A[s]);
      h[s] = h[s] * dA + (dl * bc[s]) * uv;
      acc += h[s] * bc[16 + s];
    }
    y[tk * DI + d] = acc + uv * dpv;
  }
}

// ---------------------------------------------------------------------------
// Elementwise helpers.
// ---------------------------------------------------------------------------
__global__ void add_kernel(const float* __restrict__ a, const float* __restrict__ b,
                           float* __restrict__ o, long n) {
  long i = (long)blockIdx.x * blockDim.x + threadIdx.x;
  if (i < n) o[i] = a[i] + b[i];
}

// y *= silu(res); res lives in columns [DI, 2*DI) of the in_proj output.
__global__ void ygate_kernel(float* __restrict__ y, const float* __restrict__ xr,
                             int DI, long n) {
  long i = (long)blockIdx.x * blockDim.x + threadIdx.x;
  if (i >= n) return;
  long row = i / DI; int col = (int)(i % DI);
  float r = xr[row * (long)(2 * DI) + DI + col];
  y[i] *= act_apply(r, 1);
}

// combined = concat(mamba_out, attn_out) along last dim (DM each).
__global__ void pack_kernel(const float* __restrict__ m, const float* __restrict__ a,
                            float* __restrict__ o, int DM, long n) {
  long i = (long)blockIdx.x * blockDim.x + threadIdx.x;
  if (i >= n) return;
  long row = i / (2 * DM); int c = (int)(i % (2 * DM));
  o[i] = (c < DM) ? m[row * (long)DM + c] : a[row * (long)DM + (c - DM)];
}

// x3 = x1 + attn + g*mamba + (1-g)*attn == x1 + g*mamba + (2-g)*attn  (in place)
__global__ void fuse_kernel(float* __restrict__ x1, const float* __restrict__ m,
                            const float* __restrict__ a, const float* __restrict__ g,
                            long n) {
  long i = (long)blockIdx.x * blockDim.x + threadIdx.x;
  if (i >= n) return;
  float gv = g[i];
  x1[i] = x1[i] + gv * m[i] + (2.f - gv) * a[i];
}

// ---------------------------------------------------------------------------
// Host orchestration.
// ---------------------------------------------------------------------------
static inline int cdiv_i(long a, long b) { return (int)((a + b - 1) / b); }

extern "C" void kernel_launch(void* const* d_in, const int* in_sizes, int n_in,
                              void* d_out, int out_size, void* d_ws, size_t ws_size,
                              hipStream_t stream)
{
  (void)in_sizes; (void)n_in; (void)out_size; (void)ws_size;

  const float* x        = (const float*)d_in[0];
  const float* in_proj  = (const float*)d_in[1];
  const float* conv_w   = (const float*)d_in[2];
  const float* conv_b   = (const float*)d_in[3];
  const float* x_proj   = (const float*)d_in[4];
  const float* dt_w     = (const float*)d_in[5];
  const float* dt_b     = (const float*)d_in[6];
  const float* A_log    = (const float*)d_in[7];
  const float* Dp       = (const float*)d_in[8];
  const float* out_proj = (const float*)d_in[9];
  const float* wq_w  = (const float*)d_in[10]; const float* wq_b = (const float*)d_in[11];
  const float* wk_w  = (const float*)d_in[12]; const float* wk_b = (const float*)d_in[13];
  const float* wv_w  = (const float*)d_in[14]; const float* wv_b = (const float*)d_in[15];
  const float* wo_w  = (const float*)d_in[16]; const float* wo_b = (const float*)d_in[17];
  const float* aln_g = (const float*)d_in[18]; const float* aln_b = (const float*)d_in[19];
  const float* f_w1  = (const float*)d_in[20]; const float* f_b1  = (const float*)d_in[21];
  const float* f_w2  = (const float*)d_in[22]; const float* f_b2  = (const float*)d_in[23];
  const float* n1g = (const float*)d_in[24]; const float* n1b = (const float*)d_in[25];
  const float* n2g = (const float*)d_in[26]; const float* n2b = (const float*)d_in[27];
  const float* n3g = (const float*)d_in[28]; const float* n3b = (const float*)d_in[29];
  const float* gate_w = (const float*)d_in[30]; const float* gate_b = (const float*)d_in[31];

  const int  Bsz = 8, L = 1024, DM = 512, DI = 1024, DS = 16, H = 8;
  const long T = (long)Bsz * L;                 // 8192 tokens

  // -- f32 workspace (region reuse) ----------------------------------------
  float* ws    = (float*)d_ws;
  float* xn    = ws;                            // T*512   LN1 -> xn2 -> xn3
  float* big1  = xn + T * DM;                   // T*2048  xr (xm|res) -> pv | attn_proj
  float* xc    = big1 + T * 2 * DI;             // T*1024  conv -> Q|K -> g -> ffn_h(lo)
  float* dlt   = xc + T * DI;                   // T*1024  delta -> V -> combined -> ffn_h(hi)
  float* BCb   = dlt + T * DI;                  // T*32
  float* ybuf  = BCb + T * 2 * DS;              // T*1024  scan out -> attn_out
  float* mamba = ybuf + T * DI;                 // T*512
  float* x1    = mamba + T * DM;                // T*512   x1 -> x3 (in place)

  // -- f16 packed operands --------------------------------------------------
  _Float16* h16    = (_Float16*)(x1 + T * DM);
  _Float16* wt_in  = h16;                       // [2048,512]
  _Float16* wt_xp  = wt_in  + 2048 * 512;       // [32,1024]
  _Float16* wt_dt  = wt_xp  + 32 * 1024;        // [1024,1024]
  _Float16* wt_out = wt_dt  + 1024 * 1024;      // [512,1024]
  _Float16* wt_q   = wt_out + 512 * 1024;       // [512,512]
  _Float16* wt_k   = wt_q   + 512 * 512;
  _Float16* wt_v   = wt_k   + 512 * 512;
  _Float16* wt_o   = wt_v   + 512 * 512;
  _Float16* wt_g   = wt_o   + 512 * 512;        // [512,1024]
  _Float16* wt_f1  = wt_g   + 512 * 1024;       // [2048,512]
  _Float16* wt_f2  = wt_f1  + 2048 * 512;       // [512,2048]
  _Float16* Kf16   = wt_f2  + 512 * 2048;       // [T,512]  (already N-major)
  _Float16* Vt16   = Kf16   + T * 512;          // [b,h][64][1024]

  float* xout = (float*)d_out;                  // [8,1024,512]
  float* attn = xout + T * DM;                  // [8,8,1024,1024] attn weights

  float* Q     = xc;
  float* Kmat  = xc + T * DM;
  float* V     = dlt;
  float* pv    = big1;
  float* aproj = big1 + T * DM;
  float* aout  = ybuf;
  float* comb  = dlt;
  float* g     = xc;
  float* ffh   = xc;                            // spans xc..dlt: T*2048 contiguous

  const dim3 blk(256);
  #define EW_GRID(n)  dim3((unsigned)(((n) + 255) / 256))
  #define GEMM_GRID(M, N, Z) dim3((unsigned)cdiv_i((N), 128), (unsigned)cdiv_i((M), 64), (unsigned)(Z))
  #define TR_GRID(R, C, Z)   dim3((unsigned)cdiv_i((C), 32), (unsigned)cdiv_i((R), 32), (unsigned)(Z))

  // 0. pack weights: W[K,N] f32 -> Wt[N,K] f16
  transpose_f16_kernel<<<TR_GRID(512, 2048, 1), blk, 0, stream>>>(in_proj, 2048, 0, 0, wt_in,  512,  0, 0, 512, 2048, 1);
  transpose_f16_kernel<<<TR_GRID(1024,  32, 1), blk, 0, stream>>>(x_proj,    32, 0, 0, wt_xp,  1024, 0, 0, 1024,  32, 1);
  transpose_f16_kernel<<<TR_GRID(1024,1024, 1), blk, 0, stream>>>(dt_w,    1024, 0, 0, wt_dt,  1024, 0, 0, 1024, 1024, 1);
  transpose_f16_kernel<<<TR_GRID(1024, 512, 1), blk, 0, stream>>>(out_proj, 512, 0, 0, wt_out, 1024, 0, 0, 1024,  512, 1);
  transpose_f16_kernel<<<TR_GRID(512,  512, 1), blk, 0, stream>>>(wq_w,     512, 0, 0, wt_q,   512,  0, 0, 512,  512, 1);
  transpose_f16_kernel<<<TR_GRID(512,  512, 1), blk, 0, stream>>>(wk_w,     512, 0, 0, wt_k,   512,  0, 0, 512,  512, 1);
  transpose_f16_kernel<<<TR_GRID(512,  512, 1), blk, 0, stream>>>(wv_w,     512, 0, 0, wt_v,   512,  0, 0, 512,  512, 1);
  transpose_f16_kernel<<<TR_GRID(512,  512, 1), blk, 0, stream>>>(wo_w,     512, 0, 0, wt_o,   512,  0, 0, 512,  512, 1);
  transpose_f16_kernel<<<TR_GRID(1024, 512, 1), blk, 0, stream>>>(gate_w,   512, 0, 0, wt_g,   1024, 0, 0, 1024,  512, 1);
  transpose_f16_kernel<<<TR_GRID(512, 2048, 1), blk, 0, stream>>>(f_w1,    2048, 0, 0, wt_f1,  512,  0, 0, 512, 2048, 1);
  transpose_f16_kernel<<<TR_GRID(2048, 512, 1), blk, 0, stream>>>(f_w2,     512, 0, 0, wt_f2,  2048, 0, 0, 2048,  512, 1);

  // 1. xn = LN(x, norm1)
  layernorm_kernel<<<dim3((unsigned)T), blk, 0, stream>>>(x, n1g, n1b, xn, DM);
  // 2. xr = xn @ in_proj
  gemm_wmma_kernel<<<GEMM_GRID(T, 2 * DI, 1), blk, 0, stream>>>(
      xn, DM, 0, 0, wt_in, DM, 0, 0, big1, 2 * DI, 0, 0,
      nullptr, nullptr, (int)T, 2 * DI, DM, 1, 1.f, 0);
  // 3. xc = silu(depthwise_conv(xm) + conv_b)
  conv_silu_kernel<<<EW_GRID(T * DI), blk, 0, stream>>>(
      big1, 2 * DI, conv_w, conv_b, xc, L, DI, T * DI);
  // 4. BC = xc @ x_proj
  gemm_wmma_kernel<<<GEMM_GRID(T, 2 * DS, 1), blk, 0, stream>>>(
      xc, DI, 0, 0, wt_xp, DI, 0, 0, BCb, 2 * DS, 0, 0,
      nullptr, nullptr, (int)T, 2 * DS, DI, 1, 1.f, 0);
  // 5. delta = softplus(xc @ dt_w + dt_b)
  gemm_wmma_kernel<<<GEMM_GRID(T, DI, 1), blk, 0, stream>>>(
      xc, DI, 0, 0, wt_dt, DI, 0, 0, dlt, DI, 0, 0,
      dt_b, nullptr, (int)T, DI, DI, 1, 1.f, 3);
  // 6. y = selective_scan(u=xc, delta, B, C) + u*D
  scan_kernel<<<dim3((unsigned)((Bsz * DI) / 256)), blk, 0, stream>>>(
      xc, dlt, BCb, A_log, Dp, ybuf, L, DI);
  // 7. y *= silu(res)
  ygate_kernel<<<EW_GRID(T * DI), blk, 0, stream>>>(ybuf, big1, DI, T * DI);
  // 8. mamba_out = y @ out_proj
  gemm_wmma_kernel<<<GEMM_GRID(T, DM, 1), blk, 0, stream>>>(
      ybuf, DI, 0, 0, wt_out, DI, 0, 0, mamba, DM, 0, 0,
      nullptr, nullptr, (int)T, DM, DI, 1, 1.f, 0);
  // 9. x1 = x + mamba_out
  add_kernel<<<EW_GRID(T * DM), blk, 0, stream>>>(x, mamba, x1, T * DM);
  // 10. xn2 = LN(x1, norm2)
  layernorm_kernel<<<dim3((unsigned)T), blk, 0, stream>>>(x1, n2g, n2b, xn, DM);
  // 11. Q/K/V projections
  gemm_wmma_kernel<<<GEMM_GRID(T, DM, 1), blk, 0, stream>>>(
      xn, DM, 0, 0, wt_q, DM, 0, 0, Q, DM, 0, 0,
      wq_b, nullptr, (int)T, DM, DM, 1, 1.f, 0);
  gemm_wmma_kernel<<<GEMM_GRID(T, DM, 1), blk, 0, stream>>>(
      xn, DM, 0, 0, wt_k, DM, 0, 0, Kmat, DM, 0, 0,
      wk_b, nullptr, (int)T, DM, DM, 1, 1.f, 0);
  gemm_wmma_kernel<<<GEMM_GRID(T, DM, 1), blk, 0, stream>>>(
      xn, DM, 0, 0, wt_v, DM, 0, 0, V, DM, 0, 0,
      wv_b, nullptr, (int)T, DM, DM, 1, 1.f, 0);
  // 11b. pack K -> f16 (already [S, dk]-major), V -> f16 [dk, S] per head
  convert_f16_kernel<<<EW_GRID(T * DM), blk, 0, stream>>>(Kmat, Kf16, T * DM);
  transpose_f16_kernel<<<TR_GRID(L, 64, Bsz * H), blk, 0, stream>>>(
      V, DM, (long)L * DM, 64, Vt16, L, (long)H * 64 * L, (long)64 * L, L, 64, H);
  // 12. scores = Q @ K^T / 8 per (b,h) -> d_out attn region
  gemm_wmma_kernel<<<GEMM_GRID(L, L, Bsz * H), blk, 0, stream>>>(
      Q,    DM, (long)L * DM, 64,
      Kf16, DM, (long)L * DM, 64,
      attn, L,  (long)H * L * L, (long)L * L,
      nullptr, nullptr, L, L, 64, H, 0.125f, 0);
  // 13. softmax rows in place (these ARE the attn_w outputs)
  softmax_kernel<<<dim3((unsigned)(Bsz * H * L)), blk, 0, stream>>>(attn, L);
  // 14. pv = attn_w @ V per (b,h)   (Bt = Vt16 [dk, S])
  gemm_wmma_kernel<<<GEMM_GRID(L, 64, Bsz * H), blk, 0, stream>>>(
      attn, L, (long)H * L * L, (long)L * L,
      Vt16, L, (long)H * 64 * L, (long)64 * L,
      pv,   DM, (long)L * DM, 64,
      nullptr, nullptr, L, 64, L, H, 1.f, 0);
  // 15. aproj = pv @ wo + wo_b + xn2
  gemm_wmma_kernel<<<GEMM_GRID(T, DM, 1), blk, 0, stream>>>(
      pv, DM, 0, 0, wt_o, DM, 0, 0, aproj, DM, 0, 0,
      wo_b, xn, (int)T, DM, DM, 1, 1.f, 0);
  // 16. attn_out = LN(aproj, attn_ln)
  layernorm_kernel<<<dim3((unsigned)T), blk, 0, stream>>>(aproj, aln_g, aln_b, aout, DM);
  // 17. combined = concat(mamba_out, attn_out)
  pack_kernel<<<EW_GRID(T * 2 * DM), blk, 0, stream>>>(mamba, aout, comb, DM, T * 2 * DM);
  // 18. g = sigmoid(combined @ gate_w + gate_b)
  gemm_wmma_kernel<<<GEMM_GRID(T, DM, 1), blk, 0, stream>>>(
      comb, 2 * DM, 0, 0, wt_g, 2 * DM, 0, 0, g, DM, 0, 0,
      gate_b, nullptr, (int)T, DM, 2 * DM, 1, 1.f, 4);
  // 19. x3 = x1 + attn_out + g*mamba + (1-g)*attn_out  (in place over x1)
  fuse_kernel<<<EW_GRID(T * DM), blk, 0, stream>>>(x1, mamba, aout, g, T * DM);
  // 20. xn3 = LN(x3, norm3)
  layernorm_kernel<<<dim3((unsigned)T), blk, 0, stream>>>(x1, n3g, n3b, xn, DM);
  // 21. ffn_h = gelu(xn3 @ W1 + b1)
  gemm_wmma_kernel<<<GEMM_GRID(T, 4 * DM, 1), blk, 0, stream>>>(
      xn, DM, 0, 0, wt_f1, DM, 0, 0, ffh, 4 * DM, 0, 0,
      f_b1, nullptr, (int)T, 4 * DM, DM, 1, 1.f, 2);
  // 22. x_out = x3 + ffn_h @ W2 + b2 -> d_out
  gemm_wmma_kernel<<<GEMM_GRID(T, DM, 1), blk, 0, stream>>>(
      ffh, 4 * DM, 0, 0, wt_f2, 4 * DM, 0, 0, xout, DM, 0, 0,
      f_b2, x1, (int)T, DM, 4 * DM, 1, 1.f, 0);

  #undef EW_GRID
  #undef GEMM_GRID
  #undef TR_GRID
}
